// CoLAttention_28432683499588
// MI455X (gfx1250) — matrix operand order repaired and checked by reading, not compiled, not run on hardware
//
#include <hip/hip_runtime.h>
#include <stdint.h>

// ---------------- problem constants ----------------
constexpr int Bn = 8;
constexpr int Ln = 4096;
constexpr int Dn = 1024;
constexpr int AL = 64;

constexpr int TM   = 64;    // rows (l) per workgroup
constexpr int NBLK = 256;   // e-columns staged per stage-1 block
constexpr int KC   = 64;    // d-chunk per stage-1 iteration

// ---------------- LDS layout (dynamic, 232 KB) ----------------
constexpr unsigned QS_OFF  = 0;          // [64][1024] bf16 = 128 KB
constexpr unsigned WB_OFF  = 131072;     // 2x [256][64] bf16 = 64 KB (double buffer)
constexpr unsigned XB_OFF  = 196608;     // 2x [64][64]  bf16 = 16 KB (double buffer)
constexpr unsigned AB_OFF  = 212992;     // [64][64] f32 = 16 KB
constexpr unsigned ABF_OFF = 229376;     // [64][64] bf16 = 8 KB
constexpr unsigned SMEM_BYTES = 237568;

// ---------------- vector types ----------------
typedef __attribute__((ext_vector_type(16))) __bf16   bf16x16;
typedef __attribute__((ext_vector_type(8)))  float    floatx8;
typedef __attribute__((ext_vector_type(4)))  float    floatx4;
typedef __attribute__((ext_vector_type(4)))  unsigned uintx4;
typedef __attribute__((ext_vector_type(8)))  unsigned uintx8;
typedef __attribute__((ext_vector_type(2)))  unsigned uintx2;

__device__ __forceinline__ unsigned short f2bf(float f) {
  unsigned u = __float_as_uint(f);
  u += 0x7FFFu + ((u >> 16) & 1u);      // round-to-nearest-even
  return (unsigned short)(u >> 16);
}
#if __has_builtin(__builtin_amdgcn_cvt_pk_bf16_f32)
typedef __attribute__((ext_vector_type(2))) __bf16 bf16x2;
__device__ __forceinline__ unsigned pk2(float a, float b) {
  bf16x2 r = __builtin_amdgcn_cvt_pk_bf16_f32(a, b);   // v_cvt_pk_bf16_f32
  return __builtin_bit_cast(unsigned, r);
}
#else
__device__ __forceinline__ unsigned pk2(float a, float b) {
  return (unsigned)f2bf(a) | ((unsigned)f2bf(b) << 16);
}
#endif
__device__ __forceinline__ bf16x16 mkfrag(uintx4 lo, uintx4 hi) {
  uintx8 w;
  w.s0 = lo.s0; w.s1 = lo.s1; w.s2 = lo.s2; w.s3 = lo.s3;
  w.s4 = hi.s0; w.s5 = hi.s1; w.s6 = hi.s2; w.s7 = hi.s3;
  return __builtin_bit_cast(bf16x16, w);
}
__device__ __forceinline__ floatx8 wmma_bf16(bf16x16 a, bf16x16 b, floatx8 c) {
  return __builtin_amdgcn_wmma_f32_16x16x32_bf16(false, a, false, b, (short)0, c,
                                                 false, false);
}

// ---------------- prologue: weight conversion ----------------
__global__ void cvt_wq(const float* __restrict__ src, unsigned short* __restrict__ dst) {
  int idx = (blockIdx.x * 256 + threadIdx.x) * 4;          // D*D elements
  floatx4 f = *(const floatx4*)(src + idx);
  uintx2 p; p.x = pk2(f.x, f.y); p.y = pk2(f.z, f.w);
  *(uintx2*)(dst + idx) = p;
}
__global__ void cvt_ckt(const float* __restrict__ ck, unsigned short* __restrict__ ckt) {
  int idx = blockIdx.x * 256 + threadIdx.x;                // AL*D elements, transpose
  int a = idx >> 10, d = idx & (Dn - 1);
  ckt[idx] = f2bf(ck[d * AL + a]);
}
__global__ void cvt_cv(const float* __restrict__ cv, unsigned short* __restrict__ dst) {
  int idx = blockIdx.x * 256 + threadIdx.x;                // D*AL elements
  dst[idx] = f2bf(cv[idx]);
}

// ---------------- fused main kernel ----------------
__global__ __launch_bounds__(256) void cola_main(
    const float* __restrict__ x, const unsigned char* __restrict__ mask,
    const float* __restrict__ bq,
    const unsigned short* __restrict__ WQb,   // [D][D] bf16 (row e, col d)
    const unsigned short* __restrict__ CKt,   // [AL][D] bf16 (row a, col d)
    const unsigned short* __restrict__ CVb,   // [D][AL] bf16 (row d, col a)
    float* __restrict__ out) {
  extern __shared__ char smem[];
  unsigned short* qs  = (unsigned short*)(smem + QS_OFF);
  float*          ab  = (float*)(smem + AB_OFF);
  unsigned short* abf = (unsigned short*)(smem + ABF_OFF);

  const int tid    = threadIdx.x;
  const int lane   = tid & 31;
  const int wave   = tid >> 5;
  const int laneLo = lane & 15;
  const int laneHi = lane >> 4;

  // stage-1/3 wave tiling: 2 (M) x 4 (N); each B-fragment feeds 2 WMMAs
  const int mhalf = wave & 1;
  const int nquad = wave >> 1;
  const int m0    = mhalf * 32;

  const int bidx = blockIdx.x >> 6;          // batch
  const int l0   = (blockIdx.x & 63) << 6;   // row tile base

  __builtin_prefetch(CKt, 0, 3);
  __builtin_prefetch(CVb, 0, 3);

  // cooperative-load indexing (stage 1)
  const int xrow = tid >> 2;
  const int xcg  = (tid & 3) << 4;                 // 16 floats per thread
  const unsigned wb_lds0 = (unsigned)(uintptr_t)(smem + WB_OFF);
  const unsigned long long wq_sbase = (unsigned long long)(uintptr_t)WQb;

  // ================= stage 1: Q = x * WQ^T + b =================
  for (int nb = 0; nb < 4; ++nb) {
    const int nbase = nb * NBLK;

    // double-buffered prefetch: x slice (load+cvt+ds_store) and WQ slice (async DMA)
    auto prefetch = [&](int kc, int buf) {
      const int kb = kc * KC;
      const float* xs = x + ((size_t)(bidx * Ln + l0 + xrow)) * Dn + kb + xcg;
      floatx4 f0 = *(const floatx4*)(xs + 0);
      floatx4 f1 = *(const floatx4*)(xs + 4);
      floatx4 f2 = *(const floatx4*)(xs + 8);
      floatx4 f3 = *(const floatx4*)(xs + 12);
      uintx4 p0, p1;
      p0.s0 = pk2(f0.x, f0.y); p0.s1 = pk2(f0.z, f0.w);
      p0.s2 = pk2(f1.x, f1.y); p0.s3 = pk2(f1.z, f1.w);
      p1.s0 = pk2(f2.x, f2.y); p1.s1 = pk2(f2.z, f2.w);
      p1.s2 = pk2(f3.x, f3.y); p1.s3 = pk2(f3.z, f3.w);
      unsigned short* xd =
          (unsigned short*)(smem + XB_OFF + buf * 8192 + xrow * 128) + xcg;
      *(uintx4*)(xd + 0) = p0;
      *(uintx4*)(xd + 8) = p1;

      const unsigned voff =
          (((unsigned)(nbase + tid)) * (unsigned)Dn + (unsigned)kb) * 2u;
      const unsigned lb = wb_lds0 + (unsigned)buf * 32768u + (unsigned)tid * 128u;
#pragma unroll
      for (int i = 0; i < 8; ++i) {
        asm volatile("global_load_async_to_lds_b128 %0, %1, %2"
                     :: "v"(lb + (unsigned)(i * 16)),
                        "v"(voff + (unsigned)(i * 16)),
                        "s"(wq_sbase)
                     : "memory");
      }
    };

    floatx8 acc[2][4] = {};
    prefetch(0, 0);
    for (int kc = 0; kc < Dn / KC; ++kc) {
      const int cur = kc & 1;
      asm volatile("s_wait_asynccnt 0" ::: "memory");   // my DMA into cur done
      __syncthreads();                                  // everyone's cur ready
      if (kc + 1 < Dn / KC) prefetch(kc + 1, cur ^ 1);  // overlap with compute

      const char* wbc = smem + WB_OFF + cur * 32768;
      const char* xbc = smem + XB_OFF + cur * 8192;
#pragma unroll
      for (int s = 0; s < KC; s += 32) {
        bf16x16 af[2];
#pragma unroll
        for (int mi = 0; mi < 2; ++mi) {
          const char* ar = xbc + (m0 + mi * 16 + laneLo) * 128;
          af[mi] = mkfrag(*(const uintx4*)(ar + (s + laneHi * 16) * 2),
                          *(const uintx4*)(ar + (s + 8 + laneHi * 16) * 2));
        }
#pragma unroll
        for (int j = 0; j < 4; ++j) {
          const uintx4* bp = (const uintx4*)(wbc +
              (nquad * 64 + j * 16 + laneLo) * 128 + (s + laneHi * 16) * 2);
          bf16x16 bf = mkfrag(bp[0], bp[1]);
          acc[0][j] = wmma_bf16(af[0], bf, acc[0][j]);
          acc[1][j] = wmma_bf16(af[1], bf, acc[1][j]);
        }
      }
    }
    // ---- bias + write Q tile (bf16) to LDS ----
#pragma unroll
    for (int mi = 0; mi < 2; ++mi)
#pragma unroll
      for (int j = 0; j < 4; ++j) {
        const int nglob = nbase + nquad * 64 + j * 16 + laneLo;
        const float bias = bq[nglob];
#pragma unroll
        for (int g = 0; g < 8; ++g) {
          const int row = m0 + mi * 16 + g + laneHi * 8;
          qs[row * Dn + nglob] = f2bf(acc[mi][j][g] + bias);
        }
      }
  }
  __syncthreads();   // Q tile complete

  // ================= stage 2: A = Q * C_K / 32 =================
  const int m02 = (wave & 3) * 16;   // stage-2 roles: 4 (M) x 2 (N)
  const int nh2 = wave >> 2;
  floatx8 acc2[2] = {};
#pragma unroll 2
  for (int ks = 0; ks < Dn; ks += 32) {
    const unsigned short* qr = qs + (m02 + laneLo) * Dn;
    bf16x16 afrag = mkfrag(*(const uintx4*)(qr + ks + laneHi * 16),
                           *(const uintx4*)(qr + ks + 8 + laneHi * 16));
#pragma unroll
    for (int t = 0; t < 2; ++t) {
      const int arow = nh2 * 32 + t * 16 + laneLo;
      const uintx4* bp = (const uintx4*)(CKt + arow * Dn + ks + laneHi * 16);
      acc2[t] = wmma_bf16(afrag, mkfrag(bp[0], bp[1]), acc2[t]);
    }
  }
  constexpr float kScale = 0.03125f;   // 1/sqrt(1024)
#pragma unroll
  for (int t = 0; t < 2; ++t)
#pragma unroll
    for (int g = 0; g < 8; ++g) {
      const int row = m02 + g + laneHi * 8;
      const int col = nh2 * 32 + t * 16 + laneLo;
      ab[row * AL + col] = acc2[t][g] * kScale;
    }
  __syncthreads();

  // ---- softmax over alpha (one thread per row) + mask ----
  if (tid < TM) {
    const floatx4* rp = (const floatx4*)(ab + tid * AL);
    floatx4 v[16];
    float mx = -3.402823466e38f;
#pragma unroll
    for (int i = 0; i < 16; ++i) {
      v[i] = rp[i];
      mx = fmaxf(mx, fmaxf(fmaxf(v[i].x, v[i].y), fmaxf(v[i].z, v[i].w)));
    }
    float sum = 0.f;
#pragma unroll
    for (int i = 0; i < 16; ++i) {
      v[i].x = __expf(v[i].x - mx); v[i].y = __expf(v[i].y - mx);
      v[i].z = __expf(v[i].z - mx); v[i].w = __expf(v[i].w - mx);
      sum += v[i].x + v[i].y + v[i].z + v[i].w;
    }
    const float mval = mask[bidx * Ln + l0 + tid] ? 1.f : 0.f;
    const float inv = mval / sum;
    uintx4 q0;
#pragma unroll
    for (int i = 0; i < 16; i += 2) {
      q0.s0 = pk2(v[i].x * inv, v[i].y * inv);
      q0.s1 = pk2(v[i].z * inv, v[i].w * inv);
      q0.s2 = pk2(v[i + 1].x * inv, v[i + 1].y * inv);
      q0.s3 = pk2(v[i + 1].z * inv, v[i + 1].w * inv);
      *(uintx4*)(abf + tid * AL + i * 4) = q0;
    }
  }
  __syncthreads();

  // ================= stage 3: out = A * C_V^T (2M x 4N tiling) =================
  bf16x16 a3[2][2];
#pragma unroll
  for (int mi = 0; mi < 2; ++mi) {
    const unsigned short* ar3 = abf + (m0 + mi * 16 + laneLo) * AL;
    a3[mi][0] = mkfrag(*(const uintx4*)(ar3 + laneHi * 16),
                       *(const uintx4*)(ar3 + 8 + laneHi * 16));
    a3[mi][1] = mkfrag(*(const uintx4*)(ar3 + 32 + laneHi * 16),
                       *(const uintx4*)(ar3 + 40 + laneHi * 16));
  }
#pragma unroll
  for (int blk = 0; blk < 4; ++blk) {
    const int dbase = nquad * 256 + blk * 64;
    floatx8 acc3[2][4] = {};
#pragma unroll
    for (int j = 0; j < 4; ++j) {
      const int d0 = dbase + j * 16 + laneLo;
      const uintx4* bp0 = (const uintx4*)(CVb + d0 * AL + laneHi * 16);
      bf16x16 b0 = mkfrag(bp0[0], bp0[1]);
      acc3[0][j] = wmma_bf16(a3[0][0], b0, acc3[0][j]);
      acc3[1][j] = wmma_bf16(a3[1][0], b0, acc3[1][j]);
      const uintx4* bp1 = (const uintx4*)(CVb + d0 * AL + 32 + laneHi * 16);
      bf16x16 b1 = mkfrag(bp1[0], bp1[1]);
      acc3[0][j] = wmma_bf16(a3[0][1], b1, acc3[0][j]);
      acc3[1][j] = wmma_bf16(a3[1][1], b1, acc3[1][j]);
    }
#pragma unroll
    for (int mi = 0; mi < 2; ++mi)
#pragma unroll
      for (int j = 0; j < 4; ++j) {
        const int dg = dbase + j * 16 + laneLo;
#pragma unroll
        for (int g = 0; g < 8; ++g) {
          const int row = m0 + mi * 16 + g + laneHi * 8;
          out[((size_t)(bidx * Ln + l0 + row)) * Dn + dg] = acc3[mi][j][g];
        }
      }
  }
}

// ---------------- host entry ----------------
extern "C" void kernel_launch(void* const* d_in, const int* in_sizes, int n_in,
                              void* d_out, int out_size, void* d_ws, size_t ws_size,
                              hipStream_t stream) {
  const float*         x    = (const float*)d_in[0];
  const unsigned char* mask = (const unsigned char*)d_in[1];
  const float*         wq   = (const float*)d_in[2];
  const float*         bq   = (const float*)d_in[3];
  const float*         ck   = (const float*)d_in[4];
  const float*         cv   = (const float*)d_in[5];
  float*               out  = (float*)d_out;

  char* ws = (char*)d_ws;
  unsigned short* WQb = (unsigned short*)ws;                                  // 2 MB
  unsigned short* CKt = (unsigned short*)(ws + (size_t)Dn * Dn * 2);          // 128 KB
  unsigned short* CVb = (unsigned short*)(ws + (size_t)Dn * Dn * 2 + (size_t)AL * Dn * 2);

  cvt_wq <<<(Dn * Dn) / 1024, 256, 0, stream>>>(wq, WQb);
  cvt_ckt<<<(AL * Dn) / 256, 256, 0, stream>>>(ck, CKt);
  cvt_cv <<<(Dn * AL) / 256, 256, 0, stream>>>(cv, CVb);

  (void)hipFuncSetAttribute((const void*)cola_main,
                            hipFuncAttributeMaxDynamicSharedMemorySize,
                            (int)SMEM_BYTES);
  cola_main<<<Bn * (Ln / TM), 256, SMEM_BYTES, stream>>>(x, mask, bq, WQb, CKt, CVb, out);
}
